// RQVAE_63247688401547
// MI455X (gfx1250) — matrix operand order repaired and checked by reading, not compiled, not run on hardware
//
#include <hip/hip_runtime.h>

// ---------------- problem constants ----------------
#define NQ        8       // quantizer steps
#define CDIM      512     // latent dim
#define KCB       1024    // codebook entries
#define ROWS      64      // rows per block
#define TBLK      128     // threads per block (4 wave32)
#define NTILES    64      // KCB / 16
#define NCHUNK    16      // CDIM / 32 (K-chunks per WMMA accumulation)
#define RES_STRIDE 516    // f32 stride for LDS residual rows (bank spread)
#define CBT_STRIDE 520    // bf16 stride for LDS codebook tile rows (bank spread)

typedef __attribute__((ext_vector_type(16))) __bf16        v16bf;
typedef __attribute__((ext_vector_type(8)))  float         v8f;
typedef __attribute__((ext_vector_type(16))) unsigned short v16us;
typedef __attribute__((ext_vector_type(8)))  unsigned short v8us;

__device__ __forceinline__ unsigned short f2bf(float f) {
  union { float f; unsigned int u; } c; c.f = f;
  unsigned int r = c.u + 0x7FFFu + ((c.u >> 16) & 1u);   // round-to-nearest-even
  return (unsigned short)(r >> 16);
}

// ---------------- prep: codebook f32 -> bf16, and 0.5*||cb||^2 ----------------
__global__ __launch_bounds__(256) void rq_prep(const float* __restrict__ cb,
                                               unsigned short* __restrict__ cb_bf,
                                               float* __restrict__ halfsq) {
  __shared__ float red[8];
  const int row = blockIdx.x, tid = threadIdx.x;
  const float2 v = *(const float2*)(cb + (size_t)row * CDIM + tid * 2);
  ushort2 b; b.x = f2bf(v.x); b.y = f2bf(v.y);
  *(ushort2*)(cb_bf + (size_t)row * CDIM + tid * 2) = b;
  float s = v.x * v.x + v.y * v.y;
  #pragma unroll
  for (int m = 16; m >= 1; m >>= 1) s += __shfl_xor(s, m, 32);
  if ((tid & 31) == 0) red[tid >> 5] = s;
  __syncthreads();
  if (tid == 0) {
    float t = 0.f;
    #pragma unroll
    for (int i = 0; i < 8; ++i) t += red[i];
    halfsq[row] = 0.5f * t;
  }
}

// ---------------- fused residual-quantizer ----------------
__global__ __launch_bounds__(TBLK) void rq_main(
    const float* __restrict__ z_e, const float* __restrict__ cb_f32,
    const unsigned short* __restrict__ cb_bf, const float* __restrict__ halfsq,
    float* __restrict__ out, int ntok) {
  extern __shared__ char smem[];
  float* resS         = (float*)smem;                                          // 64 x 516 f32
  unsigned short* cbS = (unsigned short*)(smem + ROWS * RES_STRIDE * 4);       // 2 x 16 x 520 bf16
  int* idxS           = (int*)(smem + ROWS * RES_STRIDE * 4 + 2 * 16 * CBT_STRIDE * 2);

  const int tid  = threadIdx.x;
  const int lane = tid & 31;
  const int wave = tid >> 5;          // 0..3, each owns 16 rows
  const int half = lane >> 4;         // 0/1 (wave32 half-group)
  const int l16  = lane & 15;
  const size_t row0 = (size_t)blockIdx.x * ROWS;

  float* residuals = out;                                   // [NQ+1, ntok, CDIM]
  float* quants    = out + (size_t)(NQ + 1) * ntok * CDIM;  // [NQ,   ntok, CDIM]

  // ---- init: z_e -> LDS residual (f32) and residuals[0] = z_e ----
  for (int it = 0; it < ROWS / 4; ++it) {
    const int r = it * 4 + wave;
    const float4* src = (const float4*)(z_e + (row0 + r) * CDIM) + lane;
    float4* dg = (float4*)(residuals + (row0 + r) * CDIM) + lane;
    float*  ds = resS + r * RES_STRIDE;
    #pragma unroll
    for (int k = 0; k < 4; ++k) {
      float4 v = src[k * 32];
      dg[k * 32] = v;
      *(float4*)(ds + (lane + k * 32) * 4) = v;
    }
  }
  __syncthreads();

  for (int q = 0; q < NQ; ++q) {
    // ---- regenerate bf16 A operands for this wave's 16 rows (register resident) ----
    // A 16x32 bf16 layout: lane L -> row M=L%16; lanes 0-15: K {0..7,16..23}+base,
    // lanes 16-31: K {8..15,24..31}+base (kbase = 8*half).
    v16bf areg[NCHUNK];
    {
      const float* myrow = resS + (wave * 16 + l16) * RES_STRIDE + half * 8;
      #pragma unroll
      for (int kc = 0; kc < NCHUNK; ++kc) {
        const float* p = myrow + kc * 32;
        float4 f0 = *(const float4*)(p);
        float4 f1 = *(const float4*)(p + 4);
        float4 g0 = *(const float4*)(p + 16);
        float4 g1 = *(const float4*)(p + 20);
        v16us u;
        u[0]=f2bf(f0.x);  u[1]=f2bf(f0.y);  u[2]=f2bf(f0.z);  u[3]=f2bf(f0.w);
        u[4]=f2bf(f1.x);  u[5]=f2bf(f1.y);  u[6]=f2bf(f1.z);  u[7]=f2bf(f1.w);
        u[8]=f2bf(g0.x);  u[9]=f2bf(g0.y);  u[10]=f2bf(g0.z); u[11]=f2bf(g0.w);
        u[12]=f2bf(g1.x); u[13]=f2bf(g1.y); u[14]=f2bf(g1.z); u[15]=f2bf(g1.w);
        areg[kc] = __builtin_bit_cast(v16bf, u);
      }
    }

    float best[8]; int bestn[8];
    #pragma unroll
    for (int j = 0; j < 8; ++j) { best[j] = -3.0e38f; bestn[j] = 0; }

    // ---- stage codebook tile 0 into LDS buffer 0 ----
    {
      const uint4* src = (const uint4*)(cb_bf) + (tid >> 3) * 64 + (tid & 7) * 8;
      uint4* dst = (uint4*)(cbS + (tid >> 3) * CBT_STRIDE + (tid & 7) * 64);
      #pragma unroll
      for (int k = 0; k < 8; ++k) dst[k] = src[k];
    }
    __syncthreads();

    for (int t = 0; t < NTILES; ++t) {
      const int cur = t & 1, nxt = cur ^ 1;
      // prefetch next tile's global data before the WMMA burst
      uint4 pre[8];
      if (t < NTILES - 1) {
        const uint4* src = (const uint4*)(cb_bf + (size_t)(t + 1) * 16 * CDIM)
                           + (tid >> 3) * 64 + (tid & 7) * 8;
        #pragma unroll
        for (int k = 0; k < 8; ++k) pre[k] = src[k];
      }

      // scores D[M][N] = sum_k res[M][k] * cb[N][k] for this 16-column tile
      v8f acc = {0.f, 0.f, 0.f, 0.f, 0.f, 0.f, 0.f, 0.f};
      // B 32x16 bf16 layout: lane L -> col N=L%16, K {0..7,16..23} + 8*half + 32*kc
      const unsigned short* bp = cbS + cur * 16 * CBT_STRIDE + l16 * CBT_STRIDE + half * 8;
      #pragma unroll
      for (int kc = 0; kc < NCHUNK; ++kc) {
        v8us blo = *(const v8us*)(bp + kc * 32);
        v8us bhi = *(const v8us*)(bp + kc * 32 + 16);
        v16us bu = __builtin_shufflevector(blo, bhi, 0,1,2,3,4,5,6,7,8,9,10,11,12,13,14,15);
        acc = __builtin_amdgcn_wmma_f32_16x16x32_bf16(
            false, areg[kc], false, __builtin_bit_cast(v16bf, bu),
            (short)0, acc, false, false);
      }

      const int n = t * 16 + l16;
      const float h = halfsq[n];
      #pragma unroll
      for (int j = 0; j < 8; ++j) {          // lane holds rows M = j + 8*half
        float s = acc[j] - h;                // argmin d == argmax (dot - 0.5||cb||^2)
        if (s > best[j]) { best[j] = s; bestn[j] = n; }
      }

      if (t < NTILES - 1) {
        uint4* dst = (uint4*)(cbS + nxt * 16 * CBT_STRIDE + (tid >> 3) * CBT_STRIDE + (tid & 7) * 64);
        #pragma unroll
        for (int k = 0; k < 8; ++k) dst[k] = pre[k];
      }
      __syncthreads();
    }

    // ---- argmin reduce across the 16 lanes of each half-group ----
    #pragma unroll
    for (int m = 1; m < 16; m <<= 1) {
      #pragma unroll
      for (int j = 0; j < 8; ++j) {
        float ov = __shfl_xor(best[j], m, 32);
        int   oi = __shfl_xor(bestn[j], m, 32);
        if (ov > best[j] || (ov == best[j] && oi < bestn[j])) { best[j] = ov; bestn[j] = oi; }
      }
    }
    if (l16 == 0) {
      #pragma unroll
      for (int j = 0; j < 8; ++j) idxS[wave * 16 + half * 8 + j] = bestn[j];
    }
    __syncthreads();

    // ---- exact f32 update: gather codeword, residual -= cw, write outputs ----
    for (int r = 0; r < 16; ++r) {
      const int idx = idxS[wave * 16 + r];
      const size_t grow = row0 + wave * 16 + r;
      const float4* cbp = (const float4*)(cb_f32 + (size_t)idx * CDIM) + lane;
      float* rs = resS + (wave * 16 + r) * RES_STRIDE;
      float4* qo = (float4*)(quants    + ((size_t)q * ntok + grow) * CDIM) + lane;
      float4* ro = (float4*)(residuals + ((size_t)(q + 1) * ntok + grow) * CDIM) + lane;
      #pragma unroll
      for (int k = 0; k < 4; ++k) {
        float4 cv = cbp[k * 32];
        float4* rp = (float4*)(rs + (lane + k * 32) * 4);
        float4 rv = *rp;
        float4 nv = make_float4(rv.x - cv.x, rv.y - cv.y, rv.z - cv.z, rv.w - cv.w);
        *rp = nv;
        qo[k * 32] = cv;
        ro[k * 32] = nv;
      }
    }
    __syncthreads();
  }
}

extern "C" void kernel_launch(void* const* d_in, const int* in_sizes, int n_in,
                              void* d_out, int out_size, void* d_ws, size_t ws_size,
                              hipStream_t stream) {
  const float* z_e = (const float*)d_in[0];
  const float* cb  = (const float*)d_in[1];
  float* out = (float*)d_out;
  const int ntok = in_sizes[0] / CDIM;   // 16384

  unsigned short* cb_bf = (unsigned short*)d_ws;                       // 1 MB bf16 codebook
  float* halfsq = (float*)((char*)d_ws + (size_t)KCB * CDIM * 2);      // 4 KB

  rq_prep<<<KCB, 256, 0, stream>>>(cb, cb_bf, halfsq);

  const size_t smem = (size_t)ROWS * RES_STRIDE * 4      // f32 residual block
                    + (size_t)2 * 16 * CBT_STRIDE * 2    // double-buffered cb tile
                    + (size_t)ROWS * 4;                  // per-row argmin indices
  (void)hipFuncSetAttribute(reinterpret_cast<const void*>(rq_main),
                            hipFuncAttributeMaxDynamicSharedMemorySize, (int)smem);
  rq_main<<<ntok / ROWS, TBLK, smem, stream>>>(z_e, cb, cb_bf, halfsq, out, ntok);
}
